// SwiGLU_4243427688667
// MI455X (gfx1250) — 8-bit matrix operand order repaired and checked by reading, compiled, not run on hardware
//
#include <hip/hip_runtime.h>
#include <stdint.h>


typedef __attribute__((ext_vector_type(8))) int v8i;

static constexpr int Bn = 4, Sn = 2048, Dn = 2048, Hn = 5632;
static constexpr int Mn = Bn * Sn;
#define EPSQ 1e-5f

__device__ __forceinline__ v8i load_a_frag(const int8_t* __restrict__ p0, int ld, int lane) {
  const int8_t* p = p0 + (size_t)(lane & 15) * ld + ((lane >> 4) << 4);
  v8i a;
#pragma unroll
  for (int c = 0; c < 4; ++c) {
    uint2 d = *reinterpret_cast<const uint2*>(p + c * 8 + (c >> 1) * 16);
    a[2 * c]     = (int)d.x;
    a[2 * c + 1] = (int)d.y;
  }
  return a;
}

__device__ __forceinline__ v8i load_b_frag(const int8_t* __restrict__ p0, int ld, int lane) {
  const int8_t* p = p0 + (size_t)(lane & 15) * ld + ((lane >> 4) << 4);
  uint4 d0 = *reinterpret_cast<const uint4*>(p);
  uint4 d1 = *reinterpret_cast<const uint4*>(p + 32);
  v8i b;
  b[0] = (int)d0.x; b[1] = (int)d0.y; b[2] = (int)d0.z; b[3] = (int)d0.w;
  b[4] = (int)d1.x; b[5] = (int)d1.y; b[6] = (int)d1.z; b[7] = (int)d1.w;
  return b;
}

__device__ __forceinline__ v8i wmma_iu8(v8i a, v8i b, v8i c) {
  return __builtin_amdgcn_wmma_i32_16x16x64_iu8(true, a, true, b, c, false, false);
}

__global__ void __launch_bounds__(256) wabs_partial_kernel(const float* __restrict__ w,
                                                           int n4, float* __restrict__ partial) {
  __shared__ float sred[256];
  const float4* w4 = reinterpret_cast<const float4*>(w);
  float s = 0.f;
  for (int j = blockIdx.x * blockDim.x + threadIdx.x; j < n4; j += gridDim.x * blockDim.x) {
    float4 v = w4[j];
    s += fabsf(v.x) + fabsf(v.y) + fabsf(v.z) + fabsf(v.w);
  }
  sred[threadIdx.x] = s;
  __syncthreads();
  for (int off = 128; off > 0; off >>= 1) {
    if (threadIdx.x < off) sred[threadIdx.x] += sred[threadIdx.x + off];
    __syncthreads();
  }
  if (threadIdx.x == 0) partial[blockIdx.x] = sred[0];
}

__global__ void __launch_bounds__(256) wfinalize_kernel(const float* __restrict__ partial,
                                                        int nparts, float inv_n,
                                                        float* __restrict__ sf) {
  __shared__ float sred[256];
  float s = 0.f;
  for (int j = threadIdx.x; j < nparts; j += 256) s += partial[j];
  sred[threadIdx.x] = s;
  __syncthreads();
  for (int off = 128; off > 0; off >>= 1) {
    if (threadIdx.x < off) sred[threadIdx.x] += sred[threadIdx.x + off];
    __syncthreads();
  }
  if (threadIdx.x == 0) {
    float f = fmaxf(sred[0] * inv_n, EPSQ);
    sf[0] = 1.f / f;
    sf[1] = f;
  }
}

__global__ void __launch_bounds__(256) quantw_kernel(const float* __restrict__ w,
                                                     const float* __restrict__ sf,
                                                     int8_t* __restrict__ wq, int n4) {
  const float s = sf[0];
  const float4* w4 = reinterpret_cast<const float4*>(w);
  char4* q4 = reinterpret_cast<char4*>(wq);
  for (int j = blockIdx.x * blockDim.x + threadIdx.x; j < n4; j += gridDim.x * blockDim.x) {
    float4 v = w4[j];
    char4 q;
    q.x = (signed char)(int)fminf(fmaxf(rintf(v.x * s), -1.f), 1.f);
    q.y = (signed char)(int)fminf(fmaxf(rintf(v.y * s), -1.f), 1.f);
    q.z = (signed char)(int)fminf(fmaxf(rintf(v.z * s), -1.f), 1.f);
    q.w = (signed char)(int)fminf(fmaxf(rintf(v.w * s), -1.f), 1.f);
    q4[j] = q;
  }
}

__global__ void __launch_bounds__(256) act_quant_x_kernel(const float* __restrict__ x,
                                                          int8_t* __restrict__ xq,
                                                          float* __restrict__ xsinv) {
  __shared__ int smax;
  if (threadIdx.x == 0) smax = 0;
  __syncthreads();
  const float4* row = reinterpret_cast<const float4*>(x + (size_t)blockIdx.x * Dn);
  float mx = 0.f;
  for (int j = threadIdx.x; j < Dn / 4; j += blockDim.x) {
    float4 v = row[j];
    mx = fmaxf(mx, fmaxf(fmaxf(fabsf(v.x), fabsf(v.y)), fmaxf(fabsf(v.z), fabsf(v.w))));
  }
  atomicMax(&smax, __float_as_int(mx));
  __syncthreads();
  const float m = fmaxf(__int_as_float(smax), EPSQ);
  const float scale = 127.f / m;
  if (threadIdx.x == 0) xsinv[blockIdx.x] = m / 127.f;
  char4* q4 = reinterpret_cast<char4*>(xq + (size_t)blockIdx.x * Dn);
  for (int j = threadIdx.x; j < Dn / 4; j += blockDim.x) {
    float4 v = row[j];
    char4 q;
    q.x = (signed char)(int)fminf(fmaxf(rintf(v.x * scale), -128.f), 127.f);
    q.y = (signed char)(int)fminf(fmaxf(rintf(v.y * scale), -128.f), 127.f);
    q.z = (signed char)(int)fminf(fmaxf(rintf(v.z * scale), -128.f), 127.f);
    q.w = (signed char)(int)fminf(fmaxf(rintf(v.w * scale), -128.f), 127.f);
    q4[j] = q;
  }
}

__global__ void __launch_bounds__(256) zero_i32_kernel(int* __restrict__ p, int n) {
  int i = blockIdx.x * blockDim.x + threadIdx.x;
  if (i < n) p[i] = 0;
}

__global__ void __launch_bounds__(256) gemm12_kernel(
    const int8_t* __restrict__ xq, const int8_t* __restrict__ w1q,
    const int8_t* __restrict__ w2q, const float* __restrict__ xsinv,
    const float* __restrict__ w1sf, const float* __restrict__ w2sf,
    _Float16* __restrict__ hbuf, int* __restrict__ rowmax) {
  __shared__ int smax[32];
  const int lane = threadIdx.x & 31;
  const int wave = threadIdx.x >> 5;
  const int m0 = blockIdx.y * 32;
  const int n0 = blockIdx.x * 256 + wave * 32;

  if (threadIdx.x < 32) smax[threadIdx.x] = 0;
  __syncthreads();

  v8i acc1[2][2], acc2[2][2];
#pragma unroll
  for (int mi = 0; mi < 2; ++mi)
#pragma unroll
    for (int ni = 0; ni < 2; ++ni) { acc1[mi][ni] = {}; acc2[mi][ni] = {}; }

  const int8_t* aptr0 = xq + (size_t)m0 * Dn;
  const int8_t* aptr1 = xq + (size_t)(m0 + 16) * Dn;
  for (int k0 = 0; k0 < Dn; k0 += 64) {
    __builtin_prefetch(aptr0 + k0 + 64, 0, 3);
    __builtin_prefetch(aptr1 + k0 + 64, 0, 3);
    v8i a0 = load_a_frag(aptr0 + k0, Dn, lane);
    v8i a1 = load_a_frag(aptr1 + k0, Dn, lane);
#pragma unroll
    for (int ni = 0; ni < 2; ++ni) {
      const size_t roff = (size_t)(n0 + 16 * ni) * Dn + k0;
      v8i b1 = load_b_frag(w1q + roff, Dn, lane);
      acc1[0][ni] = wmma_iu8(a0, b1, acc1[0][ni]);
      acc1[1][ni] = wmma_iu8(a1, b1, acc1[1][ni]);
      v8i b2 = load_b_frag(w2q + roff, Dn, lane);
      acc2[0][ni] = wmma_iu8(a0, b2, acc2[0][ni]);
      acc2[1][ni] = wmma_iu8(a1, b2, acc2[1][ni]);
    }
  }

  const float f1 = w1sf[1];
  const float f2 = w2sf[1];
  const int rsub = (lane >> 4) << 3;
  const int col0 = lane & 15;

  float xs[2][8], rmax[2][8];
#pragma unroll
  for (int mi = 0; mi < 2; ++mi)
#pragma unroll
    for (int v = 0; v < 8; ++v) {
      xs[mi][v] = xsinv[m0 + 16 * mi + rsub + v];
      rmax[mi][v] = 0.f;
    }

#pragma unroll
  for (int mi = 0; mi < 2; ++mi) {
    const int rbase = m0 + 16 * mi + rsub;
#pragma unroll
    for (int ni = 0; ni < 2; ++ni) {
      const int col = n0 + 16 * ni + col0;
#pragma unroll
      for (int v = 0; v < 8; ++v) {
        float y1 = (float)acc1[mi][ni][v] * xs[mi][v] * f1;
        float y2 = (float)acc2[mi][ni][v] * xs[mi][v] * f2;
        float g = y1 / (1.f + __expf(-y1));
        float h = g * y2;
        hbuf[(size_t)(rbase + v) * Hn + col] = (_Float16)h;
        rmax[mi][v] = fmaxf(rmax[mi][v], fabsf(h));
      }
    }
  }

#pragma unroll
  for (int mi = 0; mi < 2; ++mi)
#pragma unroll
    for (int v = 0; v < 8; ++v)
      atomicMax(&smax[16 * mi + rsub + v], __float_as_int(rmax[mi][v]));
  __syncthreads();
  if (threadIdx.x < 32) atomicMax(&rowmax[m0 + threadIdx.x], smax[threadIdx.x]);
}

__global__ void __launch_bounds__(256) hquant_kernel(const _Float16* __restrict__ hbuf,
                                                     const int* __restrict__ rowmax,
                                                     int8_t* __restrict__ hq,
                                                     float* __restrict__ hsinv) {
  const int m = blockIdx.x;
  const float mx = fmaxf(__int_as_float(rowmax[m]), EPSQ);
  const float scale = 127.f / mx;
  if (threadIdx.x == 0) hsinv[m] = mx / 127.f;
  const _Float16* row = hbuf + (size_t)m * Hn;
  char4* out4 = reinterpret_cast<char4*>(hq + (size_t)m * Hn);
  for (int j = threadIdx.x; j < Hn / 4; j += blockDim.x) {
    const _Float16* p = row + j * 4;
    char4 q;
    q.x = (signed char)(int)fminf(fmaxf(rintf((float)p[0] * scale), -128.f), 127.f);
    q.y = (signed char)(int)fminf(fmaxf(rintf((float)p[1] * scale), -128.f), 127.f);
    q.z = (signed char)(int)fminf(fmaxf(rintf((float)p[2] * scale), -128.f), 127.f);
    q.w = (signed char)(int)fminf(fmaxf(rintf((float)p[3] * scale), -128.f), 127.f);
    out4[j] = q;
  }
}

__global__ void __launch_bounds__(256) gemm3_kernel(
    const int8_t* __restrict__ hq, const int8_t* __restrict__ w3q,
    const float* __restrict__ hsinv, const float* __restrict__ w3sf,
    float* __restrict__ out) {
  const int lane = threadIdx.x & 31;
  const int wave = threadIdx.x >> 5;
  const int m0 = blockIdx.y * 32;
  const int n0 = blockIdx.x * 512 + wave * 64;

  v8i acc[2][4];
#pragma unroll
  for (int mi = 0; mi < 2; ++mi)
#pragma unroll
    for (int ni = 0; ni < 4; ++ni) acc[mi][ni] = {};

  const int8_t* aptr0 = hq + (size_t)m0 * Hn;
  const int8_t* aptr1 = hq + (size_t)(m0 + 16) * Hn;
  for (int k0 = 0; k0 < Hn; k0 += 64) {
    __builtin_prefetch(aptr0 + k0 + 64, 0, 3);
    __builtin_prefetch(aptr1 + k0 + 64, 0, 3);
    v8i a0 = load_a_frag(aptr0 + k0, Hn, lane);
    v8i a1 = load_a_frag(aptr1 + k0, Hn, lane);
#pragma unroll
    for (int ni = 0; ni < 4; ++ni) {
      v8i b = load_b_frag(w3q + (size_t)(n0 + 16 * ni) * Hn + k0, Hn, lane);
      acc[0][ni] = wmma_iu8(a0, b, acc[0][ni]);
      acc[1][ni] = wmma_iu8(a1, b, acc[1][ni]);
    }
  }

  const float f3 = w3sf[1];
  const int rsub = (lane >> 4) << 3;
  const int col0 = lane & 15;
  float hs[2][8];
#pragma unroll
  for (int mi = 0; mi < 2; ++mi)
#pragma unroll
    for (int v = 0; v < 8; ++v) hs[mi][v] = hsinv[m0 + 16 * mi + rsub + v] * f3;

#pragma unroll
  for (int mi = 0; mi < 2; ++mi) {
    const int rbase = m0 + 16 * mi + rsub;
#pragma unroll
    for (int ni = 0; ni < 4; ++ni) {
      const int col = n0 + 16 * ni + col0;
#pragma unroll
      for (int v = 0; v < 8; ++v)
        out[(size_t)(rbase + v) * Dn + col] = (float)acc[mi][ni][v] * hs[mi][v];
    }
  }
}

extern "C" void kernel_launch(void* const* d_in, const int* in_sizes, int n_in,
                              void* d_out, int out_size, void* d_ws, size_t ws_size,
                              hipStream_t stream) {
  (void)in_sizes; (void)n_in; (void)out_size; (void)ws_size;
  const float* x  = (const float*)d_in[0];
  const float* w1 = (const float*)d_in[1];
  const float* w2 = (const float*)d_in[2];
  const float* w3 = (const float*)d_in[3];
  float* out = (float*)d_out;

  uint8_t* ws = (uint8_t*)d_ws;
  size_t cur = 0;
  auto alloc = [&](size_t bytes) -> void* {
    void* p = ws + cur;
    cur = (cur + bytes + 255) & ~(size_t)255;
    return p;
  };
  int8_t*   w1q   = (int8_t*)alloc((size_t)Hn * Dn);
  int8_t*   w2q   = (int8_t*)alloc((size_t)Hn * Dn);
  int8_t*   w3q   = (int8_t*)alloc((size_t)Dn * Hn);
  int8_t*   xq    = (int8_t*)alloc((size_t)Mn * Dn);
  int8_t*   hq    = (int8_t*)alloc((size_t)Mn * Hn);
  _Float16* hbuf  = (_Float16*)alloc((size_t)Mn * Hn * 2);
  float*    xsinv = (float*)alloc((size_t)Mn * 4);
  float*    hsinv = (float*)alloc((size_t)Mn * 4);
  int*      rowmax= (int*)alloc((size_t)Mn * 4);
  float*    part1 = (float*)alloc(1024 * 4);
  float*    part2 = (float*)alloc(1024 * 4);
  float*    part3 = (float*)alloc(1024 * 4);
  float*    sf1   = (float*)alloc(2 * 4);
  float*    sf2   = (float*)alloc(2 * 4);
  float*    sf3   = (float*)alloc(2 * 4);

  const int nW  = Hn * Dn;
  const int nW4 = nW / 4;
  const float invn = 1.f / (float)nW;

  wabs_partial_kernel<<<1024, 256, 0, stream>>>(w1, nW4, part1);
  wabs_partial_kernel<<<1024, 256, 0, stream>>>(w2, nW4, part2);
  wabs_partial_kernel<<<1024, 256, 0, stream>>>(w3, nW4, part3);
  wfinalize_kernel<<<1, 256, 0, stream>>>(part1, 1024, invn, sf1);
  wfinalize_kernel<<<1, 256, 0, stream>>>(part2, 1024, invn, sf2);
  wfinalize_kernel<<<1, 256, 0, stream>>>(part3, 1024, invn, sf3);

  quantw_kernel<<<2048, 256, 0, stream>>>(w1, sf1, w1q, nW4);
  quantw_kernel<<<2048, 256, 0, stream>>>(w2, sf2, w2q, nW4);
  quantw_kernel<<<2048, 256, 0, stream>>>(w3, sf3, w3q, nW4);

  act_quant_x_kernel<<<Mn, 256, 0, stream>>>(x, xq, xsinv);

  zero_i32_kernel<<<Mn / 256, 256, 0, stream>>>(rowmax, Mn);
  gemm12_kernel<<<dim3(Hn / 256, Mn / 32), 256, 0, stream>>>(
      xq, w1q, w2q, xsinv, sf1, sf2, hbuf, rowmax);

  hquant_kernel<<<Mn, 256, 0, stream>>>(hbuf, rowmax, hq, hsinv);

  gemm3_kernel<<<dim3(Dn / 512, Mn / 32), 256, 0, stream>>>(hq, w3q, hsinv, sf3, out);
}
